// GATLayer_30769145708858
// MI455X (gfx1250) — matrix operand order repaired and checked by reading, not compiled, not run on hardware
//
#include <hip/hip_runtime.h>

typedef __attribute__((ext_vector_type(16))) _Float16 v16h;
typedef __attribute__((ext_vector_type(8)))  _Float16 v8h;
typedef __attribute__((ext_vector_type(8)))  float    v8f;

#define NB   2
#define NN   4096
#define DIN  256
#define DOUT 256
#define NE   131072
static constexpr float SCALE = 0.17677669529663687f; // 1/sqrt(32)

// ---- WMMA fragment loaders ------------------------------------------------
// A-matrix 16x32 f16: lane holds row (L&15); halves 0-15 = K c0..c0+15 contiguous,
// with c0 = 32*kk + 16*(L>=16).
__device__ inline v16h frag_a(const _Float16* p) {
  v8h lo = *(const v8h*)(p);
  v8h hh = *(const v8h*)(p + 8);
  v16h r;
#pragma unroll
  for (int t = 0; t < 8; ++t) { r[t] = lo[t]; r[t + 8] = hh[t]; }
  return r;
}
// B-matrix 32x16 f16: lane holds column (L&15); K = 16*(L>=16) + h, contiguous.
// With B stored transposed (row = out column, contiguous in K) this is one 32B load.
__device__ inline v16h frag_b(const _Float16* p) { return *(const v16h*)(p); }

__device__ inline v8f wmma(v16h a, v16h b, v8f c) {
  return __builtin_amdgcn_wmma_f32_16x16x32_f16(false, a, false, b, (short)0, c,
                                                false, false);
}

// ---- helper kernels -------------------------------------------------------
__global__ void k_zero(unsigned* __restrict__ p, int n) {
  int i = blockIdx.x * blockDim.x + threadIdx.x;
  for (; i < n; i += gridDim.x * blockDim.x) p[i] = 0u;
}

__global__ void k_scatter(const int* __restrict__ ei, unsigned* __restrict__ adj) {
  int e = blockIdx.x * blockDim.x + threadIdx.x;
  if (e < NE) {
    int s = ei[e];
    int d = ei[NE + e];
    atomicOr(&adj[(size_t)s * (NN / 32) + (d >> 5)], 1u << (d & 31));
  }
}

// Wt[w][n][k] = W_w[k][n]  (f16, transposed so B-fragments are contiguous)
__global__ void k_convw(const float* __restrict__ wq, const float* __restrict__ wk,
                        const float* __restrict__ wv, _Float16* __restrict__ wt) {
  int idx = blockIdx.x * blockDim.x + threadIdx.x; // 3*256*256
  int w = idx >> 16;
  int rem = idx & 65535;
  int n = rem >> 8, k = rem & 255;
  const float* src = (w == 0) ? wq : (w == 1) ? wk : wv;
  wt[idx] = (_Float16)src[k * 256 + n];
}

__global__ void k_convx(const float* __restrict__ x, _Float16* __restrict__ xh, int n) {
  int i = blockIdx.x * blockDim.x + threadIdx.x;
  if (i < n) xh[i] = (_Float16)x[i];
}

// ---- projection: q = x Wq, k = x Wk, v = x Wv (v stored transposed) -------
__global__ __launch_bounds__(128, 1) void k_proj(const _Float16* __restrict__ xh,
                                                 const _Float16* __restrict__ wt,
                                                 _Float16* __restrict__ qh,
                                                 _Float16* __restrict__ kh,
                                                 _Float16* __restrict__ vt) {
  const int lane = threadIdx.x & 31;
  const int wvid = threadIdx.x >> 5;
  const int rt = blockIdx.x * 4 + wvid;          // 512 row tiles of 16
  const int i0 = rt * 16;
  const int ln = lane & 15;
  const int hi = lane >> 4;
  const int aoff = hi * 16;
  const _Float16* arow = xh + (size_t)(i0 + ln) * DIN;
  v16h af[8];
#pragma unroll
  for (int kk = 0; kk < 8; ++kk) af[kk] = frag_a(arow + kk * 32 + aoff);
  const int b = i0 >> 12;
  const int jr = i0 & (NN - 1);
#pragma unroll
  for (int w = 0; w < 3; ++w) {
    const _Float16* wbase = wt + (size_t)w * DIN * DOUT;
#pragma unroll
    for (int nt = 0; nt < 16; ++nt) {
      const _Float16* brow = wbase + (size_t)(nt * 16 + ln) * DIN + hi * 16;
      v16h bf[8];                                 // one load clause ...
#pragma unroll
      for (int kk = 0; kk < 8; ++kk) bf[kk] = frag_b(brow + kk * 32);
      v8f acc = {};
#pragma unroll
      for (int kk = 0; kk < 8; ++kk) acc = wmma(af[kk], bf[kk], acc);  // ... then WMMAs
      _Float16 hv[8];
#pragma unroll
      for (int r = 0; r < 8; ++r) hv[r] = (_Float16)acc[r];  // convert first,
      if (w < 2) {                                           // then clause stores
        _Float16* dst = (w == 0) ? qh : kh;
#pragma unroll
        for (int r = 0; r < 8; ++r)
          dst[(size_t)(i0 + r + hi * 8) * DOUT + nt * 16 + ln] = hv[r];
      } else {
#pragma unroll
        for (int r = 0; r < 8; ++r)
          vt[(size_t)b * DOUT * NN + (size_t)(nt * 16 + ln) * NN + (jr + r + hi * 8)] =
              hv[r];
      }
    }
  }
}

// ---- flash-attention: one wave per 16-query tile, j-chunks of 32 ----------
__global__ __launch_bounds__(128, 1) void k_attn(const _Float16* __restrict__ qh,
                                                 const _Float16* __restrict__ kh,
                                                 const _Float16* __restrict__ vt,
                                                 const unsigned* __restrict__ adj,
                                                 float* __restrict__ out) {
  __shared__ __align__(64) _Float16 pbuf[4][16][32];
  const int lane = threadIdx.x & 31;
  const int wvid = threadIdx.x >> 5;
  const int wid = blockIdx.x * 4 + wvid;          // 512 tiles = 2 batches * 256
  const int b = wid >> 8;
  const int i0 = (wid & 255) * 16;
  const int ln = lane & 15;
  const int hi = lane >> 4;
  const int aoff = hi * 16;

  // Q fragments resident for the whole kernel (K = 256 -> 8 fragments)
  const _Float16* qrow = qh + (size_t)(b * NN + i0 + ln) * DOUT;
  v16h qf[8];
#pragma unroll
  for (int kk = 0; kk < 8; ++kk) qf[kk] = frag_a(qrow + kk * 32 + aoff);

  float m[8], l[8];
#pragma unroll
  for (int r = 0; r < 8; ++r) { m[r] = -3.0e38f; l[r] = 0.f; }
  v8f acc[16];
#pragma unroll
  for (int t = 0; t < 16; ++t) acc[t] = (v8f){};

  for (int j0 = 0; j0 < NN; j0 += 32) {
    // ---- S = Q K^T for a 16x32 chunk: two 4-deep fragment batches ----
    const _Float16* krow0 = kh + (size_t)(b * NN + j0 + ln) * DOUT + hi * 16;
    const _Float16* krow1 = krow0 + (size_t)16 * DOUT;
    v8f s0 = {}, s1 = {};
#pragma unroll
    for (int half = 0; half < 2; ++half) {
      v16h k0b[4], k1b[4];                        // 8-load clause ...
#pragma unroll
      for (int t = 0; t < 4; ++t) {
        k0b[t] = frag_b(krow0 + (half * 4 + t) * 32);
        k1b[t] = frag_b(krow1 + (half * 4 + t) * 32);
      }
#pragma unroll
      for (int t = 0; t < 4; ++t) {               // ... then 8 WMMAs (2 chains)
        s0 = wmma(qf[half * 4 + t], k0b[t], s0);
        s1 = wmma(qf[half * 4 + t], k1b[t], s1);
      }
    }
    // ---- mask + online softmax (row = r + 8*hi in C layout) ----
    const int wadr = j0 >> 5;  // 32-aligned chunk -> one adjacency word per row
#pragma unroll
    for (int r = 0; r < 8; ++r) {
      unsigned aw = adj[(size_t)(i0 + r + hi * 8) * (NN / 32) + wadr];
      float v0 = ((aw >> ln) & 1u) ? s0[r] * SCALE : -1.0e9f;
      float v1 = ((aw >> (16 + ln)) & 1u) ? s1[r] * SCALE : -1.0e9f;
      float mt = fmaxf(v0, v1);
#pragma unroll
      for (int d = 1; d < 16; d <<= 1) mt = fmaxf(mt, __shfl_xor(mt, d, 32));
      float mn = fmaxf(m[r], mt);
      float c = __expf(m[r] - mn);
      float p0 = __expf(v0 - mn);
      float p1 = __expf(v1 - mn);
      float rs = p0 + p1;
#pragma unroll
      for (int d = 1; d < 16; d <<= 1) rs += __shfl_xor(rs, d, 32);
      l[r] = l[r] * c + rs;
      m[r] = mn;
      pbuf[wvid][r + hi * 8][ln]      = (_Float16)p0;   // C-layout -> row-major stage
      pbuf[wvid][r + hi * 8][16 + ln] = (_Float16)p1;
#pragma unroll
      for (int t = 0; t < 16; ++t) acc[t][r] *= c;      // rescale O accumulator
    }
    asm volatile("s_wait_dscnt 0" ::: "memory");  // wave-local LDS transpose fence
    v16h pf = frag_a(&pbuf[wvid][ln][0] + aoff);  // P as A-operand (16x32)

    // ---- O += P V : 4-deep batches of transposed-V fragments ----
    const _Float16* vbase = vt + (size_t)b * DOUT * NN + j0 + hi * 16;
#pragma unroll
    for (int qt = 0; qt < 4; ++qt) {
      v16h vfb[4];
#pragma unroll
      for (int t = 0; t < 4; ++t)
        vfb[t] = frag_b(vbase + (size_t)((qt * 4 + t) * 16 + ln) * NN);
#pragma unroll
      for (int t = 0; t < 4; ++t)
        acc[qt * 4 + t] = wmma(pf, vfb[t], acc[qt * 4 + t]);
    }
  }
  // ---- normalize and store f32 output ----
#pragma unroll
  for (int nt = 0; nt < 16; ++nt)
#pragma unroll
    for (int r = 0; r < 8; ++r)
      out[(size_t)(b * NN + i0 + r + hi * 8) * DOUT + nt * 16 + ln] = acc[nt][r] / l[r];
}

// ---- host side ------------------------------------------------------------
extern "C" void kernel_launch(void* const* d_in, const int* in_sizes, int n_in,
                              void* d_out, int out_size, void* d_ws, size_t ws_size,
                              hipStream_t stream) {
  const float* x  = (const float*)d_in[0];
  const float* wq = (const float*)d_in[1];
  const float* wk = (const float*)d_in[2];
  const float* wv = (const float*)d_in[3];
  const int*   ei = (const int*)d_in[4];   // JAX x64 disabled -> int32
  float* out = (float*)d_out;

  char* ws = (char*)d_ws;
  unsigned*  adj = (unsigned*)(ws + 0);              // 2 MB bitmask
  _Float16*  wt  = (_Float16*)(ws + 2097152);        // 384 KB  W^T f16 (3x)
  _Float16*  xh  = (_Float16*)(ws + 2490368);        // 4 MB   x f16
  _Float16*  qh  = (_Float16*)(ws + 6684672);        // 4 MB   q f16
  _Float16*  kh  = (_Float16*)(ws + 10878976);       // 4 MB   k f16
  _Float16*  vt  = (_Float16*)(ws + 15073280);       // 4 MB   v^T f16
  (void)in_sizes; (void)n_in; (void)out_size; (void)ws_size;

  k_zero   <<<1024, 256, 0, stream>>>(adj, NN * (NN / 32));
  k_scatter<<<NE / 256, 256, 0, stream>>>(ei, adj);
  k_convw  <<<(3 * 256 * 256) / 256, 256, 0, stream>>>(wq, wk, wv, wt);
  k_convx  <<<(NB * NN * DIN) / 256, 256, 0, stream>>>(x, xh, NB * NN * DIN);
  k_proj   <<<128, 128, 0, stream>>>(xh, wt, qh, kh, vt);
  k_attn   <<<128, 128, 0, stream>>>(qh, kh, vt, adj, out);
}
